// SwinTransformerBlock_50431505990289
// MI455X (gfx1250) — matrix operand order repaired and checked by reading, not compiled, not run on hardware
//
#include <hip/hip_runtime.h>
#include <hip/hip_fp16.h>

typedef __attribute__((ext_vector_type(16))) _Float16 v16h;
typedef __attribute__((ext_vector_type(8)))  float    v8f;

#define WSZ    7
#define SHIFTV 3
#define DIMC   192
#define NHEADS 6
#define HD     32
#define NWIN   1024      // 32x32 windows of 7x7
#define NTOK   49
#define HH     224
#define SEQ    (HH*HH)   // 50176
#define ROWSB  SEQ       // per-batch rows = NWIN*NTOK = 50176
#define NBATCH 8
#define MLPH   768
#define QKVO   576
#define EPSV   1e-5f

// ---------------- WMMA fragment helpers (CDNA5 wave32 layouts) ----------------
// A 16x32 f16: lane l: m = l&15, kh = l>>4; halves 0..7 -> K = kh*8+0..7,
//                                            halves 8..15 -> K = 16+kh*8+0..7
static __device__ inline v16h load_a16(const _Float16* __restrict__ base, int stride, int lane) {
  int m = lane & 15, kh = lane >> 4;
  const _Float16* p = base + (size_t)m * stride + kh * 8;
  v16h a;
#pragma unroll
  for (int i = 0; i < 8; ++i) { a[i] = p[i]; a[8 + i] = p[16 + i]; }
  return a;
}
// B 32x16 f16: lane l: n = l&15, kh = l>>4; halves 0..7 -> K = kh*8+0..7, halves 8..15 -> K = 16+kh*8+0..7
// Source is W[n][k] row-major (B[k][n] = W[n][k]).
static __device__ inline v16h load_b16(const _Float16* __restrict__ base, int stride, int lane) {
  int n = lane & 15, kh = lane >> 4;
  const _Float16* p = base + (size_t)n * stride + kh * 8;
  v16h b;
#pragma unroll
  for (int i = 0; i < 8; ++i) { b[i] = p[i]; b[8 + i] = p[16 + i]; }
  return b;
}
#define WMMA_F16(a, b, c) \
  __builtin_amdgcn_wmma_f32_16x16x32_f16(false, (a), false, (b), (short)0, (c), false, false)

static __device__ inline float wave_sum32(float v) {
#pragma unroll
  for (int m = 16; m >= 1; m >>= 1) v += __shfl_xor(v, m, 32);
  return v;
}
static __device__ inline float half_max16(float v) {
#pragma unroll
  for (int m = 8; m >= 1; m >>= 1) v = fmaxf(v, __shfl_xor(v, m, 32));
  return v;
}
static __device__ inline float half_sum16(float v) {
#pragma unroll
  for (int m = 8; m >= 1; m >>= 1) v += __shfl_xor(v, m, 32);
  return v;
}

// ---------------- prep: weights -> f16, rel-pos bias gather ----------------
#define SZ_QKVW (QKVO*DIMC)      // 110592
#define SZ_PRJW (DIMC*DIMC)      // 36864
#define SZ_FC1W (MLPH*DIMC)      // 147456
#define SZ_FC2W (DIMC*MLPH)      // 147456
#define SZ_BIAS (NHEADS*NTOK*NTOK)

__global__ __launch_bounds__(256) void k_prep(
    const float* __restrict__ qkv_w, const float* __restrict__ proj_w,
    const float* __restrict__ fc1_w, const float* __restrict__ fc2_w,
    const float* __restrict__ rpb, const int* __restrict__ rel_idx,
    _Float16* __restrict__ wq, _Float16* __restrict__ wp,
    _Float16* __restrict__ w1, _Float16* __restrict__ w2,
    float* __restrict__ biasM) {
  int i = blockIdx.x * 256 + threadIdx.x;
  if (i < SZ_QKVW) { wq[i] = (_Float16)qkv_w[i]; return; }
  i -= SZ_QKVW;
  if (i < SZ_PRJW) { wp[i] = (_Float16)proj_w[i]; return; }
  i -= SZ_PRJW;
  if (i < SZ_FC1W) { w1[i] = (_Float16)fc1_w[i]; return; }
  i -= SZ_FC1W;
  if (i < SZ_FC2W) { w2[i] = (_Float16)fc2_w[i]; return; }
  i -= SZ_FC2W;
  if (i < SZ_BIAS) {
    int h = i / (NTOK * NTOK), rem = i % (NTOK * NTOK);
    biasM[h * (NTOK * NTOK) + rem] = rpb[rel_idx[rem] * NHEADS + h];
  }
}

// ---------------- LN1 + shift + window partition -> f16 ----------------
__global__ __launch_bounds__(256) void k_ln1(
    const float* __restrict__ x, int b,
    const float* __restrict__ g, const float* __restrict__ be,
    _Float16* __restrict__ xw) {
  int lane = threadIdx.x & 31;
  int row  = blockIdx.x * 8 + (threadIdx.x >> 5);   // window-row index
  int wi = row / NTOK, n = row % NTOK;
  int wh = wi >> 5, ww = wi & 31;
  int nh = n / WSZ, nw = n % WSZ;
  int h0 = (wh * WSZ + nh + SHIFTV) % HH;
  int w0 = (ww * WSZ + nw + SHIFTV) % HH;
  const float* src = x + ((size_t)b * SEQ + (size_t)h0 * HH + w0) * DIMC;
  float v[6]; float s = 0.f, s2 = 0.f;
#pragma unroll
  for (int i = 0; i < 6; ++i) { float t = src[i * 32 + lane]; v[i] = t; s += t; s2 += t * t; }
  s  = wave_sum32(s);
  s2 = wave_sum32(s2);
  float mean = s * (1.f / DIMC);
  float var  = s2 * (1.f / DIMC) - mean * mean;
  float rstd = rsqrtf(var + EPSV);
  _Float16* dst = xw + (size_t)row * DIMC;
#pragma unroll
  for (int i = 0; i < 6; ++i) {
    int c = i * 32 + lane;
    dst[c] = (_Float16)((v[i] - mean) * rstd * g[c] + be[c]);
  }
}

// ---------------- GEMM: qkv = xw @ qkv_w^T + b  (f16 out), 32Mx64N/wave ----------------
__global__ __launch_bounds__(256) void k_gemm_qkv(
    const _Float16* __restrict__ A, const _Float16* __restrict__ W,
    const float* __restrict__ bias, _Float16* __restrict__ out) {
  int lane = threadIdx.x & 31;
  int task = blockIdx.x * 8 + (threadIdx.x >> 5);  // 1568*9 tasks
  int mt = task / 9, ng = task % 9;
  const _Float16* ab0 = A + (size_t)mt * 32 * DIMC;
  const _Float16* ab1 = ab0 + (size_t)16 * DIMC;
  v8f z = {};
  v8f acc[2][4];
#pragma unroll
  for (int u = 0; u < 2; ++u)
#pragma unroll
    for (int j = 0; j < 4; ++j) acc[u][j] = z;
#pragma unroll
  for (int ks = 0; ks < DIMC / 32; ++ks) {
    v16h a0 = load_a16(ab0 + ks * 32, DIMC, lane);
    v16h a1 = load_a16(ab1 + ks * 32, DIMC, lane);
#pragma unroll
    for (int j = 0; j < 4; ++j) {
      v16h bf = load_b16(W + (size_t)(ng * 64 + j * 16) * DIMC + ks * 32, DIMC, lane);
      acc[0][j] = WMMA_F16(a0, bf, acc[0][j]);
      acc[1][j] = WMMA_F16(a1, bf, acc[1][j]);
    }
  }
  int kh = lane >> 4, l15 = lane & 15;
#pragma unroll
  for (int u = 0; u < 2; ++u)
#pragma unroll
    for (int j = 0; j < 4; ++j) {
      int col = ng * 64 + j * 16 + l15;
      float bb = bias[col];
#pragma unroll
      for (int r = 0; r < 8; ++r) {
        int m = mt * 32 + u * 16 + r + kh * 8;
        out[(size_t)m * QKVO + col] = (_Float16)(acc[u][j][r] + bb);
      }
    }
}

// ---------------- windowed attention, one (window, head) per block ----------------
__global__ __launch_bounds__(128) void k_attn(
    const _Float16* __restrict__ qkv, const float* __restrict__ biasM,
    const float* __restrict__ maskM, _Float16* __restrict__ ao) {
  __shared__ _Float16 sm[4][16][72];   // per-wave P strips (padded stride)
  int lane = threadIdx.x & 31;
  int wv   = threadIdx.x >> 5;         // 16-row strip index (rows padded to 64)
  int win  = blockIdx.x / NHEADS;
  int head = blockIdx.x % NHEADS;
  int kh = lane >> 4, l15 = lane & 15;

  const _Float16* qb = qkv + (size_t)win * NTOK * QKVO + head * HD;
  const _Float16* kb = qb + DIMC;
  const _Float16* vb = qb + 2 * DIMC;

  // Q A-fragment (rows wv*16.., K = 32 = full head dim)
  v16h aq;
  {
    int m = wv * 16 + l15;
    if (m < NTOK) {
      const _Float16* p = qb + (size_t)m * QKVO + kh * 8;
#pragma unroll
      for (int i = 0; i < 8; ++i) { aq[i] = p[i]; aq[8 + i] = p[16 + i]; }
    } else {
#pragma unroll
      for (int i = 0; i < 16; ++i) aq[i] = (_Float16)0.f;
    }
  }
  // S = Q * K^T  (4 column tiles)
  v8f acc[4];
#pragma unroll
  for (int nt = 0; nt < 4; ++nt) {
    v16h bk;
    int n = nt * 16 + l15;
    if (n < NTOK) {
      const _Float16* p = kb + (size_t)n * QKVO + kh * 8;
#pragma unroll
      for (int i = 0; i < 8; ++i) { bk[i] = p[i]; bk[8 + i] = p[16 + i]; }
    } else {
#pragma unroll
      for (int i = 0; i < 16; ++i) bk[i] = (_Float16)0.f;
    }
    v8f z = {};
    acc[nt] = WMMA_F16(aq, bk, z);
  }
  // scale + bias + mask, rowwise softmax (row M=r+8*kh of strip, cols across lanes)
  const float scale = 0.17677669529663687f;  // 32^-0.5
  const float* bptr = biasM + head * (NTOK * NTOK);
  const float* mptr = maskM + win * (NTOK * NTOK);
  float pv[4][8];
  float rowsum[8];
#pragma unroll
  for (int r = 0; r < 8; ++r) {
    int m = wv * 16 + r + kh * 8;
    float pm = -3.0e30f;
#pragma unroll
    for (int nt = 0; nt < 4; ++nt) {
      int n = nt * 16 + l15;
      float v;
      if (m < NTOK && n < NTOK)
        v = acc[nt][r] * scale + bptr[m * NTOK + n] + mptr[m * NTOK + n];
      else
        v = -3.0e30f;
      pv[nt][r] = v;
      pm = fmaxf(pm, v);
    }
    pm = half_max16(pm);
    float ps = 0.f;
#pragma unroll
    for (int nt = 0; nt < 4; ++nt) {
      float e = __expf(pv[nt][r] - pm);
      pv[nt][r] = e;
      ps += e;
    }
    rowsum[r] = half_sum16(ps);
  }
  // normalized P -> LDS (C-layout -> A-layout relayout)
#pragma unroll
  for (int r = 0; r < 8; ++r) {
    float inv = 1.0f / fmaxf(rowsum[r], 1e-20f);
    int rr = r + kh * 8;
    int m  = wv * 16 + rr;
#pragma unroll
    for (int nt = 0; nt < 4; ++nt) {
      float p = (m < NTOK) ? pv[nt][r] * inv : 0.f;
      sm[wv][rr][nt * 16 + l15] = (_Float16)p;
    }
  }
  __syncthreads();
  // O = P * V  (out 16x32 strip, K = 64 in 2 chunks)
#pragma unroll
  for (int nt2 = 0; nt2 < 2; ++nt2) {
    v8f oc = {};
#pragma unroll
    for (int kc = 0; kc < 2; ++kc) {
      v16h ap;
      {
        const _Float16* p = &sm[wv][l15][kc * 32 + kh * 8];
#pragma unroll
        for (int i = 0; i < 8; ++i) { ap[i] = p[i]; ap[8 + i] = p[16 + i]; }
      }
      v16h bvf;
      {
        int d = nt2 * 16 + l15;
#pragma unroll
        for (int i = 0; i < 16; ++i) {
          int key = kc * 32 + kh * 8 + (i < 8 ? i : i + 8);
          bvf[i] = (key < NTOK) ? vb[(size_t)key * QKVO + d] : (_Float16)0.f;
        }
      }
      oc = WMMA_F16(ap, bvf, oc);
    }
#pragma unroll
    for (int r = 0; r < 8; ++r) {
      int m = wv * 16 + r + kh * 8;
      if (m < NTOK)
        ao[((size_t)win * NTOK + m) * DIMC + head * HD + nt2 * 16 + l15] = (_Float16)oc[r];
    }
  }
}

// ---------------- proj GEMM + window-reverse + residual -> x1 (f32), 32Mx64N ----------------
__global__ __launch_bounds__(256) void k_gemm_proj(
    const _Float16* __restrict__ A, const _Float16* __restrict__ W,
    const float* __restrict__ bias, const float* __restrict__ x, int b,
    float* __restrict__ x1) {
  int lane = threadIdx.x & 31;
  int task = blockIdx.x * 8 + (threadIdx.x >> 5);  // 1568*3 tasks
  int mt = task / 3, ng = task % 3;
  const _Float16* ab0 = A + (size_t)mt * 32 * DIMC;
  const _Float16* ab1 = ab0 + (size_t)16 * DIMC;
  v8f z = {};
  v8f acc[2][4];
#pragma unroll
  for (int u = 0; u < 2; ++u)
#pragma unroll
    for (int j = 0; j < 4; ++j) acc[u][j] = z;
#pragma unroll
  for (int ks = 0; ks < DIMC / 32; ++ks) {
    v16h a0 = load_a16(ab0 + ks * 32, DIMC, lane);
    v16h a1 = load_a16(ab1 + ks * 32, DIMC, lane);
#pragma unroll
    for (int j = 0; j < 4; ++j) {
      v16h bf = load_b16(W + (size_t)(ng * 64 + j * 16) * DIMC + ks * 32, DIMC, lane);
      acc[0][j] = WMMA_F16(a0, bf, acc[0][j]);
      acc[1][j] = WMMA_F16(a1, bf, acc[1][j]);
    }
  }
  int kh = lane >> 4, l15 = lane & 15;
#pragma unroll
  for (int u = 0; u < 2; ++u)
#pragma unroll
    for (int j = 0; j < 4; ++j) {
      int col = ng * 64 + j * 16 + l15;
      float bb = bias[col];
#pragma unroll
      for (int r = 0; r < 8; ++r) {
        int m = mt * 32 + u * 16 + r + kh * 8;   // window-row index
        int wi = m / NTOK, n = m % NTOK;
        int wh = wi >> 5, ww = wi & 31;
        int h0 = (wh * WSZ + n / WSZ + SHIFTV) % HH;
        int w0 = (ww * WSZ + n % WSZ + SHIFTV) % HH;
        int srow = h0 * HH + w0;                 // natural seq row
        x1[(size_t)srow * DIMC + col] =
            acc[u][j][r] + bb + x[((size_t)b * SEQ + srow) * DIMC + col];
      }
    }
}

// ---------------- LN2 row stats over x1 ----------------
__global__ __launch_bounds__(256) void k_ln2stats(
    const float* __restrict__ x1, float* __restrict__ stats) {
  int lane = threadIdx.x & 31;
  int row  = blockIdx.x * 8 + (threadIdx.x >> 5);
  const float* src = x1 + (size_t)row * DIMC;
  float s = 0.f, s2 = 0.f;
#pragma unroll
  for (int i = 0; i < 6; ++i) { float t = src[i * 32 + lane]; s += t; s2 += t * t; }
  s  = wave_sum32(s);
  s2 = wave_sum32(s2);
  float mean = s * (1.f / DIMC);
  float var  = s2 * (1.f / DIMC) - mean * mean;
  if (lane == 0) {
    stats[2 * row]     = mean;
    stats[2 * row + 1] = rsqrtf(var + EPSV);
  }
}

// ---------------- fc1 GEMM (LN2 fused into A load) + GELU -> f16, 32Mx64N ----------------
__global__ __launch_bounds__(256) void k_gemm_fc1(
    const float* __restrict__ x1, const float* __restrict__ stats,
    const float* __restrict__ g, const float* __restrict__ be,
    const _Float16* __restrict__ W, const float* __restrict__ bias,
    _Float16* __restrict__ hm) {
  int lane = threadIdx.x & 31;
  int task = blockIdx.x * 8 + (threadIdx.x >> 5);  // 1568*12 tasks
  int mt = task / 12, ng = task % 12;
  int kh = lane >> 4, l15 = lane & 15;
  int mrow0 = mt * 32 + l15;
  int mrow1 = mrow0 + 16;
  float mean0 = stats[2 * mrow0], rstd0 = stats[2 * mrow0 + 1];
  float mean1 = stats[2 * mrow1], rstd1 = stats[2 * mrow1 + 1];
  const float* arow0 = x1 + (size_t)mrow0 * DIMC;
  const float* arow1 = x1 + (size_t)mrow1 * DIMC;
  v8f z = {};
  v8f acc[2][4];
#pragma unroll
  for (int u = 0; u < 2; ++u)
#pragma unroll
    for (int j = 0; j < 4; ++j) acc[u][j] = z;
#pragma unroll
  for (int ks = 0; ks < DIMC / 32; ++ks) {
    v16h a0, a1;
#pragma unroll
    for (int i = 0; i < 8; ++i) {
      int k0 = ks * 32 + kh * 8 + i;
      int k1 = k0 + 16;
      float g0 = g[k0], b0 = be[k0], g1 = g[k1], b1 = be[k1];
      a0[i]     = (_Float16)((arow0[k0] - mean0) * rstd0 * g0 + b0);
      a0[8 + i] = (_Float16)((arow0[k1] - mean0) * rstd0 * g1 + b1);
      a1[i]     = (_Float16)((arow1[k0] - mean1) * rstd1 * g0 + b0);
      a1[8 + i] = (_Float16)((arow1[k1] - mean1) * rstd1 * g1 + b1);
    }
#pragma unroll
    for (int j = 0; j < 4; ++j) {
      v16h bf = load_b16(W + (size_t)(ng * 64 + j * 16) * DIMC + ks * 32, DIMC, lane);
      acc[0][j] = WMMA_F16(a0, bf, acc[0][j]);
      acc[1][j] = WMMA_F16(a1, bf, acc[1][j]);
    }
  }
#pragma unroll
  for (int u = 0; u < 2; ++u)
#pragma unroll
    for (int j = 0; j < 4; ++j) {
      int col = ng * 64 + j * 16 + l15;
      float bb = bias[col];
#pragma unroll
      for (int r = 0; r < 8; ++r) {
        int m = mt * 32 + u * 16 + r + kh * 8;
        float h = acc[u][j][r] + bb;
        float ge = 0.5f * h * (1.0f + erff(h * 0.7071067811865475f));
        hm[(size_t)m * MLPH + col] = (_Float16)ge;
      }
    }
}

// ---------------- fc2 GEMM + residual -> d_out (f32), 32Mx64N ----------------
__global__ __launch_bounds__(256) void k_gemm_fc2(
    const _Float16* __restrict__ Hm, const _Float16* __restrict__ W,
    const float* __restrict__ bias, const float* __restrict__ x1, int b,
    float* __restrict__ out) {
  int lane = threadIdx.x & 31;
  int task = blockIdx.x * 8 + (threadIdx.x >> 5);  // 1568*3 tasks
  int mt = task / 3, ng = task % 3;
  const _Float16* ab0 = Hm + (size_t)mt * 32 * MLPH;
  const _Float16* ab1 = ab0 + (size_t)16 * MLPH;
  v8f z = {};
  v8f acc[2][4];
#pragma unroll
  for (int u = 0; u < 2; ++u)
#pragma unroll
    for (int j = 0; j < 4; ++j) acc[u][j] = z;
#pragma unroll
  for (int ks = 0; ks < MLPH / 32; ++ks) {
    v16h a0 = load_a16(ab0 + ks * 32, MLPH, lane);
    v16h a1 = load_a16(ab1 + ks * 32, MLPH, lane);
#pragma unroll
    for (int j = 0; j < 4; ++j) {
      v16h bf = load_b16(W + (size_t)(ng * 64 + j * 16) * MLPH + ks * 32, MLPH, lane);
      acc[0][j] = WMMA_F16(a0, bf, acc[0][j]);
      acc[1][j] = WMMA_F16(a1, bf, acc[1][j]);
    }
  }
  int kh = lane >> 4, l15 = lane & 15;
#pragma unroll
  for (int u = 0; u < 2; ++u)
#pragma unroll
    for (int j = 0; j < 4; ++j) {
      int col = ng * 64 + j * 16 + l15;
      float bb = bias[col];
#pragma unroll
      for (int r = 0; r < 8; ++r) {
        int m = mt * 32 + u * 16 + r + kh * 8;
        out[((size_t)b * SEQ + m) * DIMC + col] =
            acc[u][j][r] + bb + x1[(size_t)m * DIMC + col];
      }
    }
}

// ---------------- host launcher ----------------
extern "C" void kernel_launch(void* const* d_in, const int* in_sizes, int n_in,
                              void* d_out, int out_size, void* d_ws, size_t ws_size,
                              hipStream_t stream) {
  const float* x      = (const float*)d_in[0];
  const float* maskM  = (const float*)d_in[1];
  const int*   relidx = (const int*)d_in[2];
  const float* n1g    = (const float*)d_in[3];
  const float* n1b    = (const float*)d_in[4];
  const float* qkv_w  = (const float*)d_in[5];
  const float* qkv_b  = (const float*)d_in[6];
  const float* rpb    = (const float*)d_in[7];
  const float* proj_w = (const float*)d_in[8];
  const float* proj_b = (const float*)d_in[9];
  const float* n2g    = (const float*)d_in[10];
  const float* n2b    = (const float*)d_in[11];
  const float* fc1_w  = (const float*)d_in[12];
  const float* fc1_b  = (const float*)d_in[13];
  const float* fc2_w  = (const float*)d_in[14];
  const float* fc2_b  = (const float*)d_in[15];
  float* outp = (float*)d_out;

  char* ws = (char*)d_ws;
  size_t off = 0;
  auto alloc = [&](size_t bytes) -> void* {
    void* p = ws + off;
    off += (bytes + 255) & ~(size_t)255;
    return p;
  };
  _Float16* xw    = (_Float16*)alloc((size_t)ROWSB * DIMC * 2);
  _Float16* qkvb  = (_Float16*)alloc((size_t)ROWSB * QKVO * 2);
  _Float16* ao    = (_Float16*)alloc((size_t)ROWSB * DIMC * 2);
  float*    x1    = (float*)   alloc((size_t)ROWSB * DIMC * 4);
  float*    stats = (float*)   alloc((size_t)ROWSB * 2 * 4);
  _Float16* hm    = (_Float16*)alloc((size_t)ROWSB * MLPH * 2);
  _Float16* wq    = (_Float16*)alloc((size_t)SZ_QKVW * 2);
  _Float16* wp    = (_Float16*)alloc((size_t)SZ_PRJW * 2);
  _Float16* w1    = (_Float16*)alloc((size_t)SZ_FC1W * 2);
  _Float16* w2    = (_Float16*)alloc((size_t)SZ_FC2W * 2);
  float*    biasM = (float*)   alloc((size_t)SZ_BIAS * 4);

  int prep_tot = SZ_QKVW + SZ_PRJW + SZ_FC1W + SZ_FC2W + SZ_BIAS;
  k_prep<<<(prep_tot + 255) / 256, 256, 0, stream>>>(
      qkv_w, proj_w, fc1_w, fc2_w, rpb, relidx, wq, wp, w1, w2, biasM);

  for (int b = 0; b < NBATCH; ++b) {
    k_ln1<<<ROWSB / 8, 256, 0, stream>>>(x, b, n1g, n1b, xw);
    k_gemm_qkv<<<(1568 * 9) / 8, 256, 0, stream>>>(xw, wq, qkv_b, qkvb);
    k_attn<<<NWIN * NHEADS, 128, 0, stream>>>(qkvb, biasM, maskM, ao);
    k_gemm_proj<<<(1568 * 3) / 8, 256, 0, stream>>>(ao, wp, proj_b, x, b, x1);
    k_ln2stats<<<ROWSB / 8, 256, 0, stream>>>(x1, stats);
    k_gemm_fc1<<<(1568 * 12) / 8, 256, 0, stream>>>(x1, stats, n2g, n2b, w1, fc1_b, hm);
    k_gemm_fc2<<<(1568 * 3) / 8, 256, 0, stream>>>(hm, w2, fc2_b, x1, b, outp);
  }
}
